// SA_Memory_Module_5059471475315
// MI455X (gfx1250) — matrix operand order repaired and checked by reading, not compiled, not run on hardware
//
#include <hip/hip_runtime.h>
#include <hip/hip_bf16.h>
#include <cmath>

// Shapes
#define BB 8
#define CC 64
#define HH 256
#define WW 256
#define NTOK 1024
#define HWSZ 65536

typedef _Float16 half8 __attribute__((ext_vector_type(8)));
typedef _Float16 v16h  __attribute__((ext_vector_type(16)));
typedef float    v8f   __attribute__((ext_vector_type(8)));

#define WMMA_F16(a, b, c) \
  __builtin_amdgcn_wmma_f32_16x16x32_f16(false, (a), false, (b), (short)0, (c), false, false)

// A fragment: 16x32 f16, source row-major, K-contiguous, leading dim ld (halfs).
// Lane l: row r=l&15, group g=l>>4 -> halfs [8g,8g+8) and [16+8g,16+8g+8).
__device__ __forceinline__ v16h load_a_frag(const _Float16* base, int ld) {
  int lane = threadIdx.x & 31;
  int r = lane & 15, g = lane >> 4;
  const _Float16* p = base + r * ld + 8 * g;
  half8 lo = *(const half8*)p;
  half8 hi = *(const half8*)(p + 16);
  v16h out;
#pragma unroll
  for (int i = 0; i < 8; ++i) { out[i] = lo[i]; out[i + 8] = hi[i]; }
  return out;
}

// B fragment: 32x16 f16 from B^T stored [n][k] K-contiguous, leading dim ld.
// Lane l: col n=l&15, group g=l>>4 -> halfs [8g,8g+8) and [16+8g,16+8g+8) of row n.
__device__ __forceinline__ v16h load_b_frag(const _Float16* base, int ld) {
  int lane = threadIdx.x & 31;
  int n = lane & 15, g = lane >> 4;
  const _Float16* p = base + n * ld + 8 * g;
  half8 lo = *(const half8*)p;
  half8 hi = *(const half8*)(p + 16);
  v16h out;
#pragma unroll
  for (int i = 0; i < 8; ++i) { out[i] = lo[i]; out[i + 8] = hi[i]; }
  return out;
}

// ---------------- weight f32 -> f16 ----------------
// layout (halfs): wq@0 wk@4096 wv@8192 wkm@12288 wvm@16384 wz@20480 wm@36864 (total 73728)
__global__ void cvt_w_kernel(const float* __restrict__ wq, const float* __restrict__ wk,
                             const float* __restrict__ wv, const float* __restrict__ wkm,
                             const float* __restrict__ wvm, const float* __restrict__ wz,
                             const float* __restrict__ wm, _Float16* __restrict__ dst) {
  int i = blockIdx.x * 256 + threadIdx.x;
  float v;
  if      (i < 4096)  v = wq[i];
  else if (i < 8192)  v = wk[i - 4096];
  else if (i < 12288) v = wv[i - 8192];
  else if (i < 16384) v = wkm[i - 12288];
  else if (i < 20480) v = wvm[i - 16384];
  else if (i < 36864) v = wz[i - 20480];
  else                v = wm[i - 36864];
  dst[i] = (_Float16)v;
}

// ---------------- pooling: one wave per (tensor,b,c,t), mean of 64 contiguous ----------------
__global__ void pool_kernel(const float* __restrict__ h, const float* __restrict__ m,
                            _Float16* __restrict__ pth, _Float16* __restrict__ ptm) {
  int wave = threadIdx.x >> 5;
  int lane = threadIdx.x & 31;
  unsigned idx = blockIdx.x * 8u + (unsigned)wave;   // [0, 2^20)
  unsigned which = idx >> 19;
  unsigned rest  = idx & ((1u << 19) - 1u);
  unsigned b = rest >> 16;
  unsigned c = (rest >> 10) & 63u;
  unsigned t = rest & 1023u;
  const float* src = which ? m : h;
  _Float16* dst = which ? ptm : pth;
  const float* p = src + (size_t)(((b << 6) + c) << 16) + (t << 6);
  float s = p[lane] + p[lane + 32];
#pragma unroll
  for (int off = 16; off; off >>= 1) s += __shfl_xor(s, off, 32);
  if (lane == 0) dst[(((b << 10) + t) << 6) + c] = (_Float16)(s * (1.0f / 64.0f));
}

// ---------------- projections: 5 small GEMMs, one wave per 16x16 tile ----------------
// which 0:Qt 1:Kt 2:Kmt (token-major [t][c])  3:V 4:Vm (channel-major [c][t])
__global__ void proj_kernel(const _Float16* __restrict__ pth, const _Float16* __restrict__ ptm,
                            const _Float16* __restrict__ wf,
                            const float* __restrict__ bq, const float* __restrict__ bk,
                            const float* __restrict__ bkm, const float* __restrict__ bv,
                            const float* __restrict__ bvm,
                            _Float16* __restrict__ Qt, _Float16* __restrict__ Kt,
                            _Float16* __restrict__ Kmt, _Float16* __restrict__ V,
                            _Float16* __restrict__ Vm) {
  int wave = threadIdx.x >> 5, lane = threadIdx.x & 31;
  int id = blockIdx.x * 8 + wave;           // 10240 waves total
  int b = id / 1280;
  int r = id % 1280;
  int which = r >> 8;                       // 0..4
  int s = r & 255;
  int col = lane & 15, g = lane >> 4;

  const _Float16* W; const _Float16* P; const float* bias; _Float16* out;
  if      (which == 0) { W = wf + 0;     P = pth; bias = bq;  out = Qt;  }
  else if (which == 1) { W = wf + 4096;  P = pth; bias = bk;  out = Kt;  }
  else if (which == 2) { W = wf + 12288; P = ptm; bias = bkm; out = Kmt; }
  else if (which == 3) { W = wf + 8192;  P = pth; bias = bv;  out = V;   }
  else                 { W = wf + 16384; P = ptm; bias = bvm; out = Vm;  }

  if (which < 3) {
    int mt = s >> 2;  // token tile 0..63
    int nt = s & 3;   // out-channel tile 0..3
    const _Float16* Ab = P + (size_t)(((b << 10) + mt * 16) << 6);
    const _Float16* Bb = W + nt * 16 * 64;
    float bl = bias[nt * 16 + col];
    v8f acc;
#pragma unroll
    for (int v = 0; v < 8; ++v) acc[v] = bl;
#pragma unroll
    for (int kk = 0; kk < 2; ++kk)
      acc = WMMA_F16(load_a_frag(Ab + kk * 32, 64), load_b_frag(Bb + kk * 32, 64), acc);
#pragma unroll
    for (int v = 0; v < 8; ++v)
      out[(((b << 10) + mt * 16 + v + 8 * g) << 6) + nt * 16 + col] = (_Float16)acc[v];
  } else {
    int mt = s >> 6;  // channel tile 0..3
    int nt = s & 63;  // token tile 0..63
    const _Float16* Ab = W + mt * 16 * 64;
    const _Float16* Bb = P + (size_t)(((b << 10) + nt * 16) << 6);
    v8f acc;
#pragma unroll
    for (int v = 0; v < 8; ++v) acc[v] = bias[mt * 16 + 8 * g + v];
#pragma unroll
    for (int kk = 0; kk < 2; ++kk)
      acc = WMMA_F16(load_a_frag(Ab + kk * 32, 64), load_b_frag(Bb + kk * 32, 64), acc);
#pragma unroll
    for (int v = 0; v < 8; ++v)
      out[(((b << 6) + mt * 16 + v + 8 * g) << 10) + nt * 16 + col] = (_Float16)acc[v];
  }
}

// ---------------- attention: WG per (b, 16-query tile); two phases (h, m) ----------------
__global__ void attn_kernel(const _Float16* __restrict__ Qt, const _Float16* __restrict__ Kt,
                            const _Float16* __restrict__ Kmt, const _Float16* __restrict__ V,
                            const _Float16* __restrict__ Vm,
                            float* __restrict__ Zh, float* __restrict__ Zm) {
  extern __shared__ char smem[];
  float*    logits  = (float*)smem;                                   // 16*1024 f32
  _Float16* probs   = (_Float16*)(smem + 16 * 1024 * 4);              // 16*1024 f16
  float*    red     = (float*)(smem + 16 * 1024 * 4 + 16 * 1024 * 2); // 256
  float*    rowstat = red + 256;                                      // 16
  float*    zscr    = rowstat + 16;                                   // 4*256

  int tid = threadIdx.x, wave = tid >> 5, lane = tid & 31;
  int b = blockIdx.x >> 6, qt = blockIdx.x & 63;
  int col = lane & 15, g = lane >> 4;

  const _Float16* Qb = Qt + (size_t)(((b << 10) + qt * 16) << 6);
  v16h qa0 = load_a_frag(Qb, 64);
  v16h qa1 = load_a_frag(Qb + 32, 64);

  for (int ph = 0; ph < 2; ++ph) {
    const _Float16* Kp = ph ? Kmt : Kt;
    const _Float16* Vp = ph ? Vm : V;
    float* Zp = ph ? Zm : Zh;

    // logits = Q K^T  (each wave: 8 key tiles)
    for (int kt = wave; kt < 64; kt += 8) {
      const _Float16* Kb = Kp + (size_t)(((b << 10) + kt * 16) << 6);
      v8f sacc;
#pragma unroll
      for (int v = 0; v < 8; ++v) sacc[v] = 0.0f;
      sacc = WMMA_F16(qa0, load_b_frag(Kb, 64), sacc);
      sacc = WMMA_F16(qa1, load_b_frag(Kb + 32, 64), sacc);
#pragma unroll
      for (int v = 0; v < 8; ++v)
        logits[(v + 8 * g) * 1024 + kt * 16 + col] = sacc[v];
    }
    __syncthreads();

    // rowwise softmax (16 rows x 1024; 16 threads per row, 64 cols each)
    int row = tid >> 4, seg = tid & 15;
    float* lrow = logits + row * 1024 + seg * 64;
    float pm = -1e30f;
    for (int j = 0; j < 64; ++j) pm = fmaxf(pm, lrow[j]);
    red[row * 16 + seg] = pm;
    __syncthreads();
    if (seg == 0) {
      float mx = -1e30f;
      for (int j = 0; j < 16; ++j) mx = fmaxf(mx, red[row * 16 + j]);
      rowstat[row] = mx;
    }
    __syncthreads();
    float mx = rowstat[row];
    float ps = 0.0f;
    for (int j = 0; j < 64; ++j) { float e = __expf(lrow[j] - mx); lrow[j] = e; ps += e; }
    red[row * 16 + seg] = ps;
    __syncthreads();
    if (seg == 0) {
      float sm = 0.0f;
      for (int j = 0; j < 16; ++j) sm += red[row * 16 + j];
      rowstat[row] = sm;
    }
    __syncthreads();
    float inv = 1.0f / rowstat[row];
    _Float16* prow = probs + row * 1024 + seg * 64;
    for (int j = 0; j < 64; ++j) prow[j] = (_Float16)(lrow[j] * inv);
    __syncthreads();

    // Z = probs * V^T : 4 channel tiles x (2 waves split K=1024)
    int nt = wave & 3, kh = wave >> 2;
    v8f zacc;
#pragma unroll
    for (int v = 0; v < 8; ++v) zacc[v] = 0.0f;
    const _Float16* Vb = Vp + (size_t)(((b << 6) + nt * 16) << 10);
    for (int kc = 0; kc < 16; ++kc) {
      int kk = kh * 512 + kc * 32;
      zacc = WMMA_F16(load_a_frag(probs + kk, 1024), load_b_frag(Vb + kk, 1024), zacc);
    }
    if (kh == 1) {
#pragma unroll
      for (int v = 0; v < 8; ++v) zscr[nt * 256 + (v + 8 * g) * 16 + col] = zacc[v];
    }
    __syncthreads();
    if (kh == 0) {
#pragma unroll
      for (int v = 0; v < 8; ++v) {
        float z = zacc[v] + zscr[nt * 256 + (v + 8 * g) * 16 + col];
        Zp[(size_t)(((b << 10) + qt * 16 + v + 8 * g) << 6) + nt * 16 + col] = z;
      }
    }
    __syncthreads();
  }
}

// ---------------- fused upsample + wz GEMM + wm GEMM + gates; 64 pixels/WG ----------------
__global__ void fuse_kernel(const float* __restrict__ Zh, const float* __restrict__ Zm,
                            const float* __restrict__ hin, const float* __restrict__ min_,
                            const _Float16* __restrict__ wzf, const float* __restrict__ bz,
                            const _Float16* __restrict__ wmf, const float* __restrict__ bm,
                            float* __restrict__ out) {
  __shared__ __align__(16) _Float16 zcat[64][128];   // 16 KB  (B^T for GEMM1)
  __shared__ __align__(16) _Float16 cat2[64][192];   // 24 KB  (B^T for GEMM2, reused for comb)

  int tid = threadIdx.x, wave = tid >> 5, lane = tid & 31;
  int blk = blockIdx.x;
  int b = blk >> 10;
  int rem = blk & 1023;
  int y = rem >> 2;
  int x0 = (rem & 3) << 6;

  int pix = tid & 63, cg = tid >> 6;
  int x = x0 + pix;
  int colv = lane & 15, g = lane >> 4;

  // bilinear (half-pixel centers, edge clamp)
  float sy = (y + 0.5f) * 0.125f - 0.5f;
  int iy0 = (int)floorf(sy);
  float fy = sy - (float)iy0;
  int y0c = iy0 < 0 ? 0 : iy0;
  int y1c = (iy0 + 1) > 31 ? 31 : (iy0 + 1);
  float sx = (x + 0.5f) * 0.125f - 0.5f;
  int ix0 = (int)floorf(sx);
  float fx = sx - (float)ix0;
  int x0c = ix0 < 0 ? 0 : ix0;
  int x1c = (ix0 + 1) > 31 ? 31 : (ix0 + 1);
  float w00 = (1.0f - fy) * (1.0f - fx), w01 = (1.0f - fy) * fx;
  float w10 = fy * (1.0f - fx),          w11 = fy * fx;
  int t00 = ((b << 10) + y0c * 32 + x0c) << 6;
  int t01 = ((b << 10) + y0c * 32 + x1c) << 6;
  int t10 = ((b << 10) + y1c * 32 + x0c) << 6;
  int t11 = ((b << 10) + y1c * 32 + x1c) << 6;

  for (int j = 0; j < 32; ++j) {
    int ch = cg * 32 + j;
    const float* Z = (ch < 64) ? Zh : Zm;
    int cc = ch & 63;
    float val = w00 * Z[t00 + cc] + w01 * Z[t01 + cc] + w10 * Z[t10 + cc] + w11 * Z[t11 + cc];
    zcat[pix][ch] = (_Float16)val;
  }
  {
    size_t base = ((size_t)(b << 6) << 16) + (y << 8) + x;
    for (int j = 0; j < 16; ++j) {
      int c = cg * 16 + j;
      cat2[pix][128 + c] = (_Float16)hin[base + ((size_t)c << 16)];
    }
  }
  __syncthreads();

  // GEMM1: Zvec[128 x 64pix] = wz * zcat ; 32 tiles, 4 per wave, held in regs
  v8f acc1[4];
#pragma unroll
  for (int i = 0; i < 4; ++i) {
    int tt = wave + i * 8;
    int mt = tt >> 2, nt = tt & 3;
    v8f a;
#pragma unroll
    for (int v = 0; v < 8; ++v) a[v] = bz[mt * 16 + 8 * g + v];
    const _Float16* Wb = wzf + mt * 16 * 128;
    const _Float16* Bb = &zcat[nt * 16][0];
#pragma unroll
    for (int kk = 0; kk < 8; ++kk)
      a = WMMA_F16(load_a_frag(Wb + kk * 32, 128), load_b_frag(Bb + kk * 32, 128), a);
    acc1[i] = a;
  }
  __syncthreads();
#pragma unroll
  for (int i = 0; i < 4; ++i) {
    int tt = wave + i * 8;
    int mt = tt >> 2, nt = tt & 3;
    half8 hv;
#pragma unroll
    for (int v = 0; v < 8; ++v) hv[v] = (_Float16)acc1[i][v];
    *(half8*)&cat2[nt * 16 + colv][mt * 16 + 8 * g] = hv;
  }
  __syncthreads();

  // GEMM2: comb[192 x 64pix] = wm * cat2 ; 48 tiles, 6 per wave, held in regs
  v8f acc2[6];
#pragma unroll
  for (int i = 0; i < 6; ++i) {
    int tt = wave + i * 8;
    int mt = tt >> 2, nt = tt & 3;
    v8f a;
#pragma unroll
    for (int v = 0; v < 8; ++v) a[v] = bm[mt * 16 + 8 * g + v];
    const _Float16* Wb = wmf + mt * 16 * 192;
    const _Float16* Bb = &cat2[nt * 16][0];
#pragma unroll
    for (int kk = 0; kk < 6; ++kk)
      a = WMMA_F16(load_a_frag(Wb + kk * 32, 192), load_b_frag(Bb + kk * 32, 192), a);
    acc2[i] = a;
  }
  __syncthreads();
#pragma unroll
  for (int i = 0; i < 6; ++i) {
    int tt = wave + i * 8;
    int mt = tt >> 2, nt = tt & 3;
    half8 hv;
#pragma unroll
    for (int v = 0; v < 8; ++v) hv[v] = (_Float16)acc2[i][v];
    *(half8*)&cat2[nt * 16 + colv][mt * 16 + 8 * g] = hv;
  }
  __syncthreads();

  // gates
  {
    size_t base = ((size_t)(b << 6) << 16) + (y << 8) + x;
    for (int j = 0; j < 16; ++j) {
      int c = cg * 16 + j;
      float mo = (float)cat2[pix][c];
      float mg = (float)cat2[pix][64 + c];
      float mi = (float)cat2[pix][128 + c];
      float mv = min_[base + ((size_t)c << 16)];
      float si = 1.0f / (1.0f + __expf(-mi));
      float nm = (1.0f - si) * mv + si * tanhf(mg);
      float nh = (1.0f / (1.0f + __expf(-mo))) * nm;
      out[base + ((size_t)c << 16)] = nh;
      out[(size_t)33554432 + base + ((size_t)c << 16)] = nm;
    }
  }
}

extern "C" void kernel_launch(void* const* d_in, const int* in_sizes, int n_in,
                              void* d_out, int out_size, void* d_ws, size_t ws_size,
                              hipStream_t stream) {
  const float* h   = (const float*)d_in[0];
  const float* m   = (const float*)d_in[1];
  const float* wq  = (const float*)d_in[2];
  const float* bq  = (const float*)d_in[3];
  const float* wk  = (const float*)d_in[4];
  const float* bk  = (const float*)d_in[5];
  const float* wv  = (const float*)d_in[6];
  const float* bv  = (const float*)d_in[7];
  const float* wkm = (const float*)d_in[8];
  const float* bkm = (const float*)d_in[9];
  const float* wvm = (const float*)d_in[10];
  const float* bvm = (const float*)d_in[11];
  const float* wz  = (const float*)d_in[12];
  const float* bz  = (const float*)d_in[13];
  const float* wm  = (const float*)d_in[14];
  const float* bm  = (const float*)d_in[15];
  float* out = (float*)d_out;

  char* ws = (char*)d_ws;
  const size_t MB = 1024 * 1024;
  _Float16* pth = (_Float16*)(ws + 0 * MB);
  _Float16* ptm = (_Float16*)(ws + 1 * MB);
  _Float16* Qt  = (_Float16*)(ws + 2 * MB);
  _Float16* Kt  = (_Float16*)(ws + 3 * MB);
  _Float16* Kmt = (_Float16*)(ws + 4 * MB);
  _Float16* V   = (_Float16*)(ws + 5 * MB);
  _Float16* Vm  = (_Float16*)(ws + 6 * MB);
  float*    Zh  = (float*)(ws + 7 * MB);
  float*    Zm  = (float*)(ws + 9 * MB);
  _Float16* wf  = (_Float16*)(ws + 11 * MB);

  cvt_w_kernel<<<288, 256, 0, stream>>>(wq, wk, wv, wkm, wvm, wz, wm, wf);
  pool_kernel<<<131072, 256, 0, stream>>>(h, m, pth, ptm);
  proj_kernel<<<1280, 256, 0, stream>>>(pth, ptm, wf, bq, bk, bkm, bv, bvm,
                                        Qt, Kt, Kmt, V, Vm);

  size_t smem = 16 * 1024 * 4 + 16 * 1024 * 2 + (256 + 16 + 1024) * 4;
  hipFuncSetAttribute((const void*)attn_kernel,
                      hipFuncAttributeMaxDynamicSharedMemorySize, (int)smem);
  attn_kernel<<<512, 256, smem, stream>>>(Qt, Kt, Kmt, V, Vm, Zh, Zm);

  fuse_kernel<<<8192, 256, 0, stream>>>(Zh, Zm, h, m, wf + 20480, bz, wf + 36864, bm, out);
}
